// SeqDecoderLSTM_31988916420802
// MI455X (gfx1250) — matrix operand order repaired and checked by reading, not compiled, not run on hardware
//
#include <hip/hip_runtime.h>
#include <hip/hip_bf16.h>

// ---------------------------------------------------------------------------
// Types for CDNA5 WMMA (wave32): v_wmma_f32_16x16x32_bf16
// ---------------------------------------------------------------------------
typedef __bf16 bf16_t;
typedef bf16_t bf16x8  __attribute__((ext_vector_type(8)));
typedef bf16_t bf16x16 __attribute__((ext_vector_type(16)));
typedef float  f32x8   __attribute__((ext_vector_type(8)));

// A-matrix 16x32 bf16 fragment (ISA 7.12.2 table):
//   lanes 0-15 : row M=lane,   K = {0..7} then {16..23}
//   lanes16-31 : row M=lane-16,K = {8..15} then {24..31}
__device__ __forceinline__ bf16x16 load_a_frag(const bf16_t* __restrict__ base,
                                               int ld, int row0, int k0, int lane) {
  const int l  = lane & 15;
  const int hl = lane >> 4;
  const bf16_t* p = base + (size_t)(row0 + l) * ld + k0 + hl * 8;
  bf16x8 lo = *(const bf16x8*)(p);        // K chunk {0..7}  (+8*hl)
  bf16x8 hi = *(const bf16x8*)(p + 16);   // K chunk {16..23}(+8*hl)
  return __builtin_shufflevector(lo, hi, 0,1,2,3,4,5,6,7,8,9,10,11,12,13,14,15);
}

// B-matrix 32x16 bf16 fragment (same K order as A fragment):
//   lanes 0-15 : column N=lane,    K = {0..7} then {16..23}
//   lanes16-31 : column N=lane-16, K = {8..15} then {24..31}
// W is stored row-major [N,K], so column N of op(B) is row N of W.
__device__ __forceinline__ bf16x16 load_b_frag(const bf16_t* __restrict__ W,
                                               int ld, int n0, int k0, int lane) {
  const int l  = lane & 15;
  const int hl = lane >> 4;
  const bf16_t* p = W + (size_t)(n0 + l) * ld + k0 + hl * 8;
  bf16x8 lo = *(const bf16x8*)(p);
  bf16x8 hi = *(const bf16x8*)(p + 16);
  return __builtin_shufflevector(lo, hi, 0,1,2,3,4,5,6,7,8,9,10,11,12,13,14,15);
}

__device__ __forceinline__ f32x8 wmma_bf16(bf16x16 a, bf16x16 b, f32x8 c) {
  return __builtin_amdgcn_wmma_f32_16x16x32_bf16(
      /*neg_a=*/false, a, /*neg_b=*/false, b,
      /*c_mod=*/(short)0, c, /*reuse_a=*/false, /*reuse_b=*/false);
}

// ---------------------------------------------------------------------------
// GEMM: C[M,N] = A[M,K] (bf16) * W[N,K]^T (bf16) + bias[N]  (fp32 out)
// Block = 128 threads (4 waves). Block tile 128x64; each wave does 32(M)x64(N)
// = 8 WMMAs per K-chunk. Ping-pong double buffering with two NAMED buffer
// sets (manual unroll by 2) so no register-rotation copies are needed:
//   load buf1(k+32); wmma(buf0); [load buf0(k+64)]; wmma(buf1)
// M%128==0, N%64==0, K%64==0 assumed (holds: 1024, 2048, 512).
// ---------------------------------------------------------------------------
__global__ __launch_bounds__(128)
void wmma_gemm_bt_kernel(const bf16_t* __restrict__ A,
                         const bf16_t* __restrict__ W,
                         const float*  __restrict__ bias,
                         float* __restrict__ C,
                         int M, int N, int K) {
  const int lane   = threadIdx.x & 31;
  const int wave   = threadIdx.x >> 5;
  const int tile_n = blockIdx.x * 64;
  const int tile_m = blockIdx.y * 128 + wave * 32;
  (void)M;

  f32x8 acc[2][4];
#pragma unroll
  for (int mi = 0; mi < 2; ++mi)
#pragma unroll
    for (int j = 0; j < 4; ++j) acc[mi][j] = f32x8{};

  bf16x16 A0[2], B0[4];   // ping buffer
  bf16x16 A1[2], B1[4];   // pong buffer

  // Prologue: fragments for k = 0 into buffer 0.
  A0[0] = load_a_frag(A, K, tile_m,      0, lane);
  A0[1] = load_a_frag(A, K, tile_m + 16, 0, lane);
#pragma unroll
  for (int j = 0; j < 4; ++j) B0[j] = load_b_frag(W, K, tile_n + 16 * j, 0, lane);

  for (int k = 0; k < K; k += 64) {
    // Stage buffer 1 with chunk k+32 (always in range since K % 64 == 0).
    A1[0] = load_a_frag(A, K, tile_m,      k + 32, lane);
    A1[1] = load_a_frag(A, K, tile_m + 16, k + 32, lane);
#pragma unroll
    for (int j = 0; j < 4; ++j)
      B1[j] = load_b_frag(W, K, tile_n + 16 * j, k + 32, lane);

    // Speculative prefetch hints two chunks ahead (dropped if OOB).
    __builtin_prefetch(A + (size_t)(tile_m + (lane & 15)) * K + k + 96, 0, 1);
    __builtin_prefetch(W + (size_t)(tile_n + (lane & 15)) * K + k + 96, 0, 1);

    // Consume buffer 0 (chunk k); buffer-1 loads remain in flight.
#pragma unroll
    for (int mi = 0; mi < 2; ++mi)
#pragma unroll
      for (int j = 0; j < 4; ++j)
        acc[mi][j] = wmma_bf16(A0[mi], B0[j], acc[mi][j]);

    // Stage buffer 0 with chunk k+64 (uniform guard, last iteration skips).
    if (k + 64 < K) {
      A0[0] = load_a_frag(A, K, tile_m,      k + 64, lane);
      A0[1] = load_a_frag(A, K, tile_m + 16, k + 64, lane);
#pragma unroll
      for (int j = 0; j < 4; ++j)
        B0[j] = load_b_frag(W, K, tile_n + 16 * j, k + 64, lane);
    }

    // Consume buffer 1 (chunk k+32).
#pragma unroll
    for (int mi = 0; mi < 2; ++mi)
#pragma unroll
      for (int j = 0; j < 4; ++j)
        acc[mi][j] = wmma_bf16(A1[mi], B1[j], acc[mi][j]);
  }

  // C/D layout: VGPR r -> lanes 0-15: (M=r, N=lane); lanes 16-31: (M=r+8, N=lane-16)
  const int l  = lane & 15;
  const int hl = lane >> 4;
#pragma unroll
  for (int mi = 0; mi < 2; ++mi) {
#pragma unroll
    for (int j = 0; j < 4; ++j) {
      const int n = tile_n + 16 * j + l;
      const float bv = bias ? bias[n] : 0.0f;
#pragma unroll
      for (int r = 0; r < 8; ++r) {
        const int m = tile_m + 16 * mi + r + 8 * hl;
        C[(size_t)m * N + n] = acc[mi][j][r] + bv;
      }
    }
  }
}

// ---------------------------------------------------------------------------
// fp32 -> bf16 conversion (grid-stride)
// ---------------------------------------------------------------------------
__global__ void f32_to_bf16_kernel(const float* __restrict__ in,
                                   bf16_t* __restrict__ out, long n) {
  long i = (long)blockIdx.x * blockDim.x + threadIdx.x;
  long stride = (long)gridDim.x * blockDim.x;
  for (; i < n; i += stride) out[i] = (bf16_t)in[i];
}

// ---------------------------------------------------------------------------
// Transpose W_ih [4H,V] -> W_ihT [V,4H] so per-batch one-hot gathers become
// coalesced row reads in the cell kernel. Read coalesced, write strided.
// ---------------------------------------------------------------------------
__global__ void transpose_kernel(const float* __restrict__ in,
                                 float* __restrict__ out, int R, int Ccol) {
  long n = (long)R * Ccol;
  long i = (long)blockIdx.x * blockDim.x + threadIdx.x;
  long stride = (long)gridDim.x * blockDim.x;
  for (; i < n; i += stride) {
    int r = (int)(i / Ccol);
    int c = (int)(i % Ccol);
    out[(size_t)c * R + r] = in[i];
  }
}

// ---------------------------------------------------------------------------
// Init: h0(bf16), c0 copy, idx=-1, mask=1, bsum = b_ih + b_hh
// ---------------------------------------------------------------------------
__global__ void init_state_kernel(const float* __restrict__ eh,
                                  const float* __restrict__ ec,
                                  const float* __restrict__ bih,
                                  const float* __restrict__ bhh,
                                  bf16_t* __restrict__ h, float* __restrict__ c,
                                  int* __restrict__ idx, float* __restrict__ mask,
                                  float* __restrict__ bsum,
                                  long BH, int B, int H4) {
  long i = (long)blockIdx.x * blockDim.x + threadIdx.x;
  long stride = (long)gridDim.x * blockDim.x;
  for (; i < BH; i += stride) {
    h[i] = (bf16_t)eh[i];
    c[i] = ec[i];
    if (i < B)  { idx[i] = -1; mask[i] = 1.0f; }
    if (i < H4) { bsum[i] = bih[i] + bhh[i]; }
  }
}

// ---------------------------------------------------------------------------
// x0gates[k] = sum_v init_input[v] * W_ih[k, v]   (block-per-row reduction)
// ---------------------------------------------------------------------------
__global__ __launch_bounds__(256)
void x0gates_kernel(const float* __restrict__ W_ih,
                    const float* __restrict__ x0,
                    float* __restrict__ out, int V) {
  const int k = blockIdx.x;
  const float* row = W_ih + (size_t)k * V;
  float s = 0.0f;
  for (int v = threadIdx.x; v < V; v += 256) s += row[v] * x0[v];
  __shared__ float red[256];
  red[threadIdx.x] = s;
  __syncthreads();
  for (int st = 128; st > 0; st >>= 1) {
    if (threadIdx.x < st) red[threadIdx.x] += red[threadIdx.x + st];
    __syncthreads();
  }
  if (threadIdx.x == 0) out[k] = red[0];
}

// ---------------------------------------------------------------------------
// LSTM cell (elementwise). gates holds h@W_hh^T only; this adds the one-hot
// input contribution (row idx[b] of W_ihT, or x0gates at t=0) + bias.
// Torch gate order i,f,g,o at offsets 0,H,2H,3H.
// ---------------------------------------------------------------------------
__global__ __launch_bounds__(256)
void lstm_cell_kernel(const float* __restrict__ gates,   // [B,4H]
                      const float* __restrict__ bsum,    // [4H]
                      const float* __restrict__ W_ihT,   // [V,4H]
                      const float* __restrict__ x0g,     // [4H]
                      const int*   __restrict__ idx,     // [B]
                      float* __restrict__ c,             // [B,H]
                      bf16_t* __restrict__ h,            // [B,H] bf16
                      int B, int H, int V) {
  (void)V;
  int t = blockIdx.x * blockDim.x + threadIdx.x;
  if (t >= B * H) return;
  const int b = t / H;
  const int j = t % H;
  const int id = idx[b];
  const int H4 = 4 * H;
  const float* g = gates + (size_t)b * H4;
  const float* xrow = (id < 0) ? x0g : (W_ihT + (size_t)id * H4);  // coalesced

  const float gi = g[j]         + bsum[j]         + xrow[j];
  const float gf = g[H + j]     + bsum[H + j]     + xrow[H + j];
  const float gg = g[2 * H + j] + bsum[2 * H + j] + xrow[2 * H + j];
  const float go = g[3 * H + j] + bsum[3 * H + j] + xrow[3 * H + j];

  const float si = 1.0f / (1.0f + __expf(-gi));
  const float sf = 1.0f / (1.0f + __expf(-gf));
  const float so = 1.0f / (1.0f + __expf(-go));
  const float tg = tanhf(gg);

  const float cn = sf * c[t] + si * tg;
  const float hn = so * tanhf(cn);
  c[t] = cn;
  h[t] = (bf16_t)hn;
}

// ---------------------------------------------------------------------------
// Argmax over V per batch row + one-hot predicts + mask bookkeeping.
// masks output is the PRE-update mask (matches jax.lax.scan output).
// First-occurrence tie-break like jnp.argmax.
// ---------------------------------------------------------------------------
__global__ __launch_bounds__(256)
void argmax_onehot_kernel(const float* __restrict__ logits,   // [B,V] (this t)
                          float* __restrict__ predicts,       // [B,V] (this t)
                          float* __restrict__ masks_out,      // [B]   (this t)
                          float* __restrict__ mask,           // [B] state
                          int*   __restrict__ idx,            // [B] state
                          const int* __restrict__ eos_idx_p,
                          int V) {
  const int b = blockIdx.x;
  const float* row = logits + (size_t)b * V;

  float best = -3.402823466e+38f;
  int   bi   = 0x7fffffff;
  for (int v = threadIdx.x; v < V; v += 256) {
    float x = row[v];
    if (x > best) { best = x; bi = v; }   // ascending v => first occurrence
  }
  __shared__ float sval[256];
  __shared__ int   sidx[256];
  sval[threadIdx.x] = best;
  sidx[threadIdx.x] = bi;
  __syncthreads();
  for (int s = 128; s > 0; s >>= 1) {
    if (threadIdx.x < s) {
      float v2 = sval[threadIdx.x + s];
      int   i2 = sidx[threadIdx.x + s];
      if (v2 > sval[threadIdx.x] ||
          (v2 == sval[threadIdx.x] && i2 < sidx[threadIdx.x])) {
        sval[threadIdx.x] = v2;
        sidx[threadIdx.x] = i2;
      }
    }
    __syncthreads();
  }
  const int amax = sidx[0];

  if (threadIdx.x == 0) {
    masks_out[b] = mask[b];                 // pre-update mask is emitted
    if (amax == *eos_idx_p) mask[b] = 0.0f; // mask *= (1 - onehot[eos])
    idx[b] = amax;                          // next x = one-hot(amax)
  }
  float* prow = predicts + (size_t)b * V;
  for (int v = threadIdx.x; v < V; v += 256)
    prow[v] = (v == amax) ? 1.0f : 0.0f;
}

// ---------------------------------------------------------------------------
// Host launcher. Inputs (setup_inputs order):
//  0 encoder_hidden [B,H]  1 encoder_cell [B,H]
//  2 W_ih [4H,V]  3 W_hh [4H,H]  4 b_ih [4H]  5 b_hh [4H]
//  6 W_out [V,H]  7 b_out [V]  8 init_input [1,V]  9 max_len  10 eos_idx
// d_out = predicts[T,B,V] | logits[T,B,V] | masks[T,1,B]  (fp32)
// ---------------------------------------------------------------------------
extern "C" void kernel_launch(void* const* d_in, const int* in_sizes, int n_in,
                              void* d_out, int out_size, void* d_ws, size_t ws_size,
                              hipStream_t stream) {
  (void)n_in; (void)ws_size;
  const float* eh    = (const float*)d_in[0];
  const float* ec    = (const float*)d_in[1];
  const float* W_ih  = (const float*)d_in[2];
  const float* W_hh  = (const float*)d_in[3];
  const float* b_ih  = (const float*)d_in[4];
  const float* b_hh  = (const float*)d_in[5];
  const float* W_out = (const float*)d_in[6];
  const float* b_out = (const float*)d_in[7];
  const float* x0    = (const float*)d_in[8];
  const int*   eos   = (const int*)d_in[10];

  // Derive dims from sizes: b_ih = 4H, W_out = V*H, encoder_hidden = B*H.
  const int H  = in_sizes[4] / 4;
  const int V  = in_sizes[6] / H;
  const int B  = in_sizes[0] / H;
  const int H4 = 4 * H;
  const long BV  = (long)B * V;
  const long BH  = (long)B * H;
  const int  T   = (int)((long)out_size / (2 * BV + B));

  // Workspace carve-out (256B aligned).
  char* w = (char*)d_ws;
  size_t off = 0;
  auto carve = [&](size_t bytes) -> void* {
    void* p = w + off;
    off = (off + bytes + 255) & ~(size_t)255;
    return p;
  };
  bf16_t* whh_bf  = (bf16_t*)carve((size_t)H4 * H * sizeof(bf16_t));
  bf16_t* wout_bf = (bf16_t*)carve((size_t)V * H * sizeof(bf16_t));
  float*  wihT    = (float*)carve((size_t)V * H4 * sizeof(float));
  bf16_t* h_bf    = (bf16_t*)carve((size_t)BH * sizeof(bf16_t));
  float*  c_st    = (float*)carve((size_t)BH * sizeof(float));
  float*  gates   = (float*)carve((size_t)B * H4 * sizeof(float));
  float*  bsum    = (float*)carve((size_t)H4 * sizeof(float));
  float*  x0g     = (float*)carve((size_t)H4 * sizeof(float));
  int*    idx     = (int*)carve((size_t)B * sizeof(int));
  float*  mask    = (float*)carve((size_t)B * sizeof(float));

  float* out = (float*)d_out;
  float* predicts_base = out;
  float* logits_base   = out + (size_t)T * BV;
  float* masks_base    = out + 2 * (size_t)T * BV;

  // --- Prep (once per launch; fully re-done each call => deterministic) ---
  {
    const long n1 = (long)H4 * H;
    const long n2 = (long)V * H;
    f32_to_bf16_kernel<<<1024, 256, 0, stream>>>(W_hh, whh_bf, n1);
    f32_to_bf16_kernel<<<1024, 256, 0, stream>>>(W_out, wout_bf, n2);
    transpose_kernel<<<2048, 256, 0, stream>>>(W_ih, wihT, H4, V);
    init_state_kernel<<<1024, 256, 0, stream>>>(eh, ec, b_ih, b_hh, h_bf, c_st,
                                                idx, mask, bsum, BH, B, H4);
    x0gates_kernel<<<H4, 256, 0, stream>>>(W_ih, x0, x0g, V);
  }

  // --- Sequential decode loop ---
  const dim3 gemm_block(128);
  const dim3 gemm1_grid(H4 / 64, B / 128);  // gates: [B,4H]
  const dim3 gemm2_grid(V / 64, B / 128);   // logits: [B,V]
  const int cell_blocks = (int)((BH + 255) / 256);

  for (int t = 0; t < T; ++t) {
    // gates = h @ W_hh^T   (bias + one-hot input added in cell kernel)
    wmma_gemm_bt_kernel<<<gemm1_grid, gemm_block, 0, stream>>>(
        h_bf, whh_bf, nullptr, gates, B, H4, H);

    // elementwise LSTM cell -> c (fp32), h (bf16)
    lstm_cell_kernel<<<cell_blocks, 256, 0, stream>>>(
        gates, bsum, wihT, x0g, idx, c_st, h_bf, B, H, V);

    // logits = h @ W_out^T + b_out  (written straight into d_out)
    float* logits_t = logits_base + (size_t)t * BV;
    wmma_gemm_bt_kernel<<<gemm2_grid, gemm_block, 0, stream>>>(
        h_bf, wout_bf, b_out, logits_t, B, V, H);

    // argmax -> one-hot predicts, masks, next idx/mask
    argmax_onehot_kernel<<<B, 256, 0, stream>>>(
        logits_t, predicts_base + (size_t)t * BV, masks_base + (size_t)t * B,
        mask, idx, eos, V);
  }
}
